// ScaledDotAttention_28750511079451
// MI455X (gfx1250) — compile-verified, matrix operand order checked by reading, not run on hardware
//
#include <hip/hip_runtime.h>

typedef __attribute__((ext_vector_type(16))) _Float16 v16h;
typedef __attribute__((ext_vector_type(8)))  _Float16 v8h;
typedef __attribute__((ext_vector_type(4)))  _Float16 v4h;
typedef __attribute__((ext_vector_type(8)))  float    v8f;

constexpr int   B = 2, L = 2048, S = 2048, H = 16, E = 64;
constexpr int   QS = H * E;                 // row stride (floats) in q/k/v
constexpr int   W  = S / 32;                // mask words per row
constexpr float SCALE   = 0.125f;           // 1/sqrt(64)
constexpr float LOG_EPS = -18.420680743952367f;   // log(1e-8)

__device__ __forceinline__ v8f wmma16(v16h a, v16h b, v8f c) {
  return __builtin_amdgcn_wmma_f32_16x16x32_f16(false, a, false, b, (short)0, c,
                                                false, false);
}

// Load 16 f16 elements covering k = kbase..kbase+7 and 16+kbase..16+kbase+7
// from a row of f32 (global or LDS) -- the 16-bit WMMA A/B per-lane striping.
__device__ __forceinline__ v16h load_tile16(const float* rowp, int kbase) {
  const float4 f0 = *(const float4*)(rowp + kbase);
  const float4 f1 = *(const float4*)(rowp + kbase + 4);
  const float4 f2 = *(const float4*)(rowp + 16 + kbase);
  const float4 f3 = *(const float4*)(rowp + 16 + kbase + 4);
  v16h r;
  r[0]  = (_Float16)f0.x; r[1]  = (_Float16)f0.y; r[2]  = (_Float16)f0.z; r[3]  = (_Float16)f0.w;
  r[4]  = (_Float16)f1.x; r[5]  = (_Float16)f1.y; r[6]  = (_Float16)f1.z; r[7]  = (_Float16)f1.w;
  r[8]  = (_Float16)f2.x; r[9]  = (_Float16)f2.y; r[10] = (_Float16)f2.z; r[11] = (_Float16)f2.w;
  r[12] = (_Float16)f3.x; r[13] = (_Float16)f3.y; r[14] = (_Float16)f3.z; r[15] = (_Float16)f3.w;
  return r;
}

// ---- kernel 0: rowok[l] = 0 if hard_mask row l is all zeros, else 1 ----
__global__ __launch_bounds__(256)
void rowmask_kernel(const float* __restrict__ hm, float* __restrict__ rowok) {
  __shared__ float red[256];
  const int l = blockIdx.x;
  float s = 0.f;
  for (int c = threadIdx.x; c < S; c += 256) s += hm[(size_t)l * S + c];
  red[threadIdx.x] = s;
  __syncthreads();
  for (int o = 128; o > 0; o >>= 1) {
    if (threadIdx.x < o) red[threadIdx.x] += red[threadIdx.x + o];
    __syncthreads();
  }
  if (threadIdx.x == 0) rowok[l] = (red[0] == 0.f) ? 0.f : 1.f;
}

// ---- kernel 1: packed allow-bits per (b, row, 32-key word):
// bit = !(causal && pos_s > pos_q) && (hm != 0 || row_all_masked) ----
__global__ __launch_bounds__(256)
void maskbits_kernel(const float* __restrict__ hm, const float* __restrict__ pos,
                     const float* __restrict__ rowok, const int* __restrict__ CAUSAL,
                     unsigned* __restrict__ mbits) {
  const int wid = blockIdx.x * 256 + threadIdx.x;      // B*L*W words
  if (wid >= B * L * W) return;
  const int w = wid % W;
  const int l = (wid / W) % L;
  const int b = wid / (W * L);
  const bool  causal = (CAUSAL[0] != 0);
  const float pq = pos[(size_t)b * L + l];
  const bool  am = (rowok[l] == 0.f);
  unsigned bits = 0u;
  const int s0 = w * 32;
#pragma unroll 4
  for (int j = 0; j < 32; ++j) {
    const int s = s0 + j;
    bool ok = true;
    if (causal && pos[(size_t)b * L + s] > pq) ok = false;
    if (!am && hm[(size_t)l * S + s] == 0.f) ok = false;
    bits |= (ok ? 1u : 0u) << j;
  }
  mbits[wid] = bits;
}

// ---- main fused attention kernel ----
__global__ __launch_bounds__(256)
void attn_fused(const float* __restrict__ Q, const float* __restrict__ K,
                const float* __restrict__ V, const float* __restrict__ ROWOK,
                const unsigned* __restrict__ MBITS,
                float* __restrict__ outV, float* __restrict__ outAtt,
                float* __restrict__ outEnt)
{
  // padded strides chosen for conflict-free 16B ds loads across 16 lanes
  __shared__ __align__(16) _Float16 lk [32][72];   // K chunk (32 keys x 64 dims), 4.5 KB
  __shared__ __align__(16) _Float16 lvT[64][40];   // V chunk transposed [dim][key], 5 KB
  __shared__ __align__(16) float    pbuf[8][16][36]; // per-wave p tiles (f32), 18 KB

  const int tid   = threadIdx.x;
  const int wave  = tid >> 5;
  const int lane  = tid & 31;
  const int halfw = lane >> 4;
  const int lan16 = lane & 15;
  const int kbase = halfw * 8;

  const int b  = blockIdx.z;
  const int h  = blockIdx.y;
  const int q0 = blockIdx.x * 128 + wave * 16;

  const float* Qb = Q + ((size_t)b * L * H + h) * E;
  const float* Kb = K + ((size_t)b * S * H + h) * E;
  const float* Vb = V + ((size_t)b * S * H + h) * E;
  float*     attb = outAtt + (size_t)(b * H + h) * L * S;

  // Q tile (16x64) -> f16 WMMA A layout, lives in registers for both passes
  const float* qrowp = Qb + (size_t)(q0 + lan16) * QS;
  const v16h aq0 = load_tile16(qrowp, kbase);
  const v16h aq1 = load_tile16(qrowp + 32, kbase);

  int    qrow[8];
  float  zf[8];
  size_t mb[8];
#pragma unroll
  for (int i = 0; i < 8; ++i) {
    qrow[i] = q0 + i + 8 * halfw;
    zf[i]   = ROWOK[qrow[i]];
    mb[i]   = ((size_t)b * L + qrow[i]) * W;
  }

  // cooperative stage: K chunk -> lk (coalesced global, packed b64 ds stores)
  auto stage_k = [&](int s0) {
#pragma unroll
    for (int j = 0; j < 2; ++j) {
      const int fidx = tid + 256 * j;           // 512 float4s
      const int key  = fidx >> 4;               // coalesced 256B per key row
      const int dseg = (fidx & 15) * 4;
      const float4 kv = *(const float4*)(Kb + (size_t)(s0 + key) * QS + dseg);
      v4h pk;
      pk[0] = (_Float16)kv.x; pk[1] = (_Float16)kv.y;
      pk[2] = (_Float16)kv.z; pk[3] = (_Float16)kv.w;
      *(v4h*)(&lk[key][dseg]) = pk;
    }
  };
  // cooperative stage: V chunk -> lvT transposed
  auto stage_v = [&](int s0) {
#pragma unroll
    for (int j = 0; j < 2; ++j) {
      const int fidx = tid + 256 * j;
      const int key  = fidx & 31;               // L2 gather, conflict-free ds scatter
      const int dseg = (fidx >> 5) * 4;
      const float4 vv = *(const float4*)(Vb + (size_t)(s0 + key) * QS + dseg);
      lvT[dseg + 0][key] = (_Float16)vv.x;
      lvT[dseg + 1][key] = (_Float16)vv.y;
      lvT[dseg + 2][key] = (_Float16)vv.z;
      lvT[dseg + 3][key] = (_Float16)vv.w;
    }
  };

  // masked scaled score tile t (16 keys) of the current staged 32-key chunk
  auto score_tile = [&](int t, const unsigned (&mw)[8], float (&xs)[8]) {
    const v8h* kp = (const v8h*)(&lk[t * 16 + lan16][0]);
    const v8h r0 = kp[halfw], r1 = kp[2 + halfw], r2 = kp[4 + halfw], r3 = kp[6 + halfw];
    v16h bk0, bk1;
#pragma unroll
    for (int e = 0; e < 8; ++e) {
      bk0[e] = r0[e]; bk0[8 + e] = r1[e];
      bk1[e] = r2[e]; bk1[8 + e] = r3[e];
    }
    v8f c = {};
    c = wmma16(aq0, bk0, c);
    c = wmma16(aq1, bk1, c);
    const int sh = t * 16 + lan16;
#pragma unroll
    for (int i = 0; i < 8; ++i)
      xs[i] = ((mw[i] >> sh) & 1u) ? c[i] * SCALE : -__builtin_inff();
  };

  // ---------------- pass A: online row (max, sum exp), one exp per element ----
  float m[8], ssum[8];
#pragma unroll
  for (int i = 0; i < 8; ++i) { m[i] = -3.0e38f; ssum[i] = 0.f; }

  for (int s0 = 0; s0 < S; s0 += 32) {
    __syncthreads();
    stage_k(s0);
    __syncthreads();
    unsigned mw[8];
#pragma unroll
    for (int i = 0; i < 8; ++i) mw[i] = MBITS[mb[i] + (s0 >> 5)];
#pragma unroll
    for (int t = 0; t < 2; ++t) {
      float xs[8];
      score_tile(t, mw, xs);
#pragma unroll
      for (int i = 0; i < 8; ++i) {
        const float d = xs[i] - m[i];
        const float e = __expf(-__builtin_fabsf(d));      // exp(-|x-m|): 1 trans
        ssum[i] = (xs[i] > m[i]) ? __builtin_fmaf(ssum[i], e, 1.f) : (ssum[i] + e);
        m[i] = fmaxf(m[i], xs[i]);
      }
    }
  }
  // merge the 16 per-lane partials of each row
#pragma unroll
  for (int off = 1; off < 16; off <<= 1) {
#pragma unroll
    for (int i = 0; i < 8; ++i) {
      const float mo = __shfl_xor(m[i], off, 16);
      const float so = __shfl_xor(ssum[i], off, 16);
      const float mn = fmaxf(m[i], mo);
      ssum[i] = ssum[i] * __expf(m[i] - mn) + so * __expf(mo - mn);
      m[i] = mn;
    }
  }
  float logs[8], invz[8];
#pragma unroll
  for (int i = 0; i < 8; ++i) {
    logs[i] = __logf(ssum[i]);
    invz[i] = (1.0f / ssum[i]) * zf[i];       // fold all-masked zeroing into 1/s
  }

  // ------- pass B: normalize, write att, entropy, accumulate O = att @ V -------
  float ent[8];
#pragma unroll
  for (int i = 0; i < 8; ++i) ent[i] = 0.f;
  v8f acc0 = {}, acc1 = {}, acc2 = {}, acc3 = {};

  for (int s0 = 0; s0 < S; s0 += 32) {
    __syncthreads();
    stage_k(s0);
    stage_v(s0);
    __syncthreads();
    unsigned mw[8];
#pragma unroll
    for (int i = 0; i < 8; ++i) mw[i] = MBITS[mb[i] + (s0 >> 5)];
#pragma unroll
    for (int t = 0; t < 2; ++t) {
      float xs[8];
      score_tile(t, mw, xs);
#pragma unroll
      for (int i = 0; i < 8; ++i) {
        const float d = xs[i] - m[i];
        const float p = __expf(d) * invz[i];
        ent[i] += p * fmaxf(d - logs[i], LOG_EPS);        // p * log(clip(p,1e-8))
        pbuf[wave][i + 8 * halfw][t * 16 + lan16] = p;
      }
    }
    asm volatile("s_wait_dscnt 0" ::: "memory");          // same-wave LDS RAW fence

    // att: each lane stores a contiguous 64B row segment (4x b128)
    {
      const float* pr = &pbuf[wave][lan16][halfw * 16];
      float* ob = attb + (size_t)(q0 + lan16) * S + s0 + halfw * 16;
      const float4 a0 = *(const float4*)(pr + 0);
      const float4 a1 = *(const float4*)(pr + 4);
      const float4 a2 = *(const float4*)(pr + 8);
      const float4 a3 = *(const float4*)(pr + 12);
      *(float4*)(ob + 0)  = a0;
      *(float4*)(ob + 4)  = a1;
      *(float4*)(ob + 8)  = a2;
      *(float4*)(ob + 12) = a3;
    }
    // att tile -> WMMA A layout (same striping helper, reading LDS f32)
    const v16h ap = load_tile16(&pbuf[wave][lan16][0], kbase);

    // four 32x16 V fragments from lvT, accumulate 16x64 output tile
#pragma unroll
    for (int n = 0; n < 4; ++n) {
      const v8h* vrow = (const v8h*)(&lvT[n * 16 + lan16][0]);
      const v8h blo = vrow[halfw];
      const v8h bhi = vrow[2 + halfw];
      v16h bv;
#pragma unroll
      for (int e = 0; e < 8; ++e) { bv[e] = blo[e]; bv[8 + e] = bhi[e]; }
      if      (n == 0) acc0 = wmma16(ap, bv, acc0);
      else if (n == 1) acc1 = wmma16(ap, bv, acc1);
      else if (n == 2) acc2 = wmma16(ap, bv, acc2);
      else             acc3 = wmma16(ap, bv, acc3);
    }
  }

  // entropy: sum across the 16 lanes of each half, one lane per row writes
#pragma unroll
  for (int off = 1; off < 16; off <<= 1) {
#pragma unroll
    for (int i = 0; i < 8; ++i) ent[i] += __shfl_xor(ent[i], off, 16);
  }
  float* entb = outEnt + (size_t)(b * H + h) * L;
#pragma unroll
  for (int i = 0; i < 8; ++i)
    if (lan16 == i) entb[qrow[i]] = -ent[i];

  // O tile
#pragma unroll
  for (int i = 0; i < 8; ++i) {
    float* op = outV + ((size_t)(b * L + qrow[i]) * H + h) * E + lan16;
    op[0]  = acc0[i];
    op[16] = acc1[i];
    op[32] = acc2[i];
    op[48] = acc3[i];
  }
}

extern "C" void kernel_launch(void* const* d_in, const int* in_sizes, int n_in,
                              void* d_out, int out_size, void* d_ws, size_t ws_size,
                              hipStream_t stream) {
  (void)in_sizes; (void)n_in; (void)out_size; (void)ws_size;
  const float* q      = (const float*)d_in[0];
  const float* k      = (const float*)d_in[1];
  const float* v      = (const float*)d_in[2];
  // d_in[3], d_in[4] (mask_miss_k/q) unused by the reference
  const float* pos    = (const float*)d_in[5];
  const float* hm     = (const float*)d_in[6];
  const int*   causal = (const int*)d_in[7];

  float* out    = (float*)d_out;
  float* outV   = out;                                    // B*L*H*E
  float* outAtt = outV + (size_t)B * L * H * E;           // B*H*L*S
  float* outEnt = outAtt + (size_t)B * H * L * S;         // B*H*L

  float*    rowok = (float*)d_ws;                         // L floats
  unsigned* mbits = (unsigned*)((char*)d_ws + 8192);      // B*L*W u32 (1 MB)

  rowmask_kernel<<<L, 256, 0, stream>>>(hm, rowok);
  maskbits_kernel<<<(B * L * W + 255) / 256, 256, 0, stream>>>(hm, pos, rowok,
                                                               causal, mbits);
  dim3 grid(L / 128, H, B);
  attn_fused<<<grid, 256, 0, stream>>>(q, k, v, rowok, mbits,
                                       outV, outAtt, outEnt);
}
